// Model_83691732730406
// MI455X (gfx1250) — matrix operand order repaired and checked by reading, not compiled, not run on hardware
//
#include <hip/hip_runtime.h>
#include <hip/hip_bf16.h>

typedef _Float16 half_t;
typedef __attribute__((ext_vector_type(16))) _Float16 v16h;
typedef __attribute__((ext_vector_type(8)))  _Float16 v8h;
typedef __attribute__((ext_vector_type(8)))  float    v8f;
typedef __attribute__((ext_vector_type(4)))  unsigned int v4u;
typedef __attribute__((ext_vector_type(8)))  int      v8i_;
typedef __attribute__((ext_vector_type(4)))  int      v4i_;

constexpr int NB = 512;    // batch
constexpr int NS = 168;    // encoder sequence length
constexpr int NI = 8;      // input features (padded to 32 for WMMA K)
constexpr int NH = 512;    // hidden size
constexpr int NF = 256;    // fc1 width
constexpr int NP = 24;     // decoder steps
constexpr int G4 = 4 * NH; // 2048 gate outputs

constexpr int BT = 32;     // batch rows per block (2 row-groups of 16)

#if defined(__HIP_DEVICE_COMPILE__) && __has_builtin(__builtin_amdgcn_tensor_load_to_lds)
#define HAVE_TDM 1
#else
#define HAVE_TDM 0
#endif

// ---------------------------------------------------------------------------
// WMMA fragment loaders — wave32 16x16x32 f16 fragments in this code's contiguous k order (k = 16h + s, A and B alike).
// A (16x32, MxK): lanes 0-15 hold row m=lane, K = kc+[0..15];
//                 lanes 16-31 hold row m=lane-16, K = kc+[16..31].
// B (32x16, KxN): lane holds column n=lane&15; lanes 0-15 K=kc+[0..15],
//                 lanes 16-31 K=kc+[16..31] (16 contiguous halves -> b128 pair).
// Works for both global and LDS pointers (addrspace inferred after inlining;
// LDS path lowers to ds_load_b128).
// ---------------------------------------------------------------------------
__device__ inline v16h load_a_frag(const half_t* X, int ldx, int m0, int kc) {
  const int lane = threadIdx.x & 31;
  const int m    = lane & 15;
  const int koff = (lane & 16) ? 16 : 0;
  const half_t* p = X + (size_t)(m0 + m) * ldx + kc + koff;
  v8h lo = *(const v8h*)(p);
  v8h hi = *(const v8h*)(p + 8);
  v16h a;
#pragma unroll
  for (int i = 0; i < 8; ++i) { a[i] = lo[i]; a[i + 8] = hi[i]; }
  return a;
}

__device__ inline v16h load_b_frag(const half_t* W, int ldw, int n0, int kc) {
  const int lane = threadIdx.x & 31;
  const int n    = lane & 15;
  const int kb   = kc + ((lane & 16) ? 16 : 0);
  const half_t* p = W + (size_t)(n0 + n) * ldw + kb;
  v8h lo = *(const v8h*)(p);
  v8h hi = *(const v8h*)(p + 8);
  v16h b;
#pragma unroll
  for (int i = 0; i < 8; ++i) { b[i] = lo[i]; b[i + 8] = hi[i]; }
  return b;
}

__device__ inline float sigm(float v) { return 1.0f / (1.0f + __expf(-v)); }

#define WMMA_F16(A, Bf, C) \
  __builtin_amdgcn_wmma_f32_16x16x32_f16(false, (A), false, (Bf), (short)0, (C), false, false)

// ---------------------------------------------------------------------------
// Issue one TDM descriptor: contiguous 1-D tile (n8 x 8 bytes) global -> LDS
// offset 0 (the kernel's single __shared__ array). CDNA5 ISA §8: group0 =
// {count=1 | lds_addr | global_addr | type=2}; group1 = {data_size=3(8B),
// tensor_dim0=n8, tensor_dim1=1, tile_dim0=n8, tile_dim1=1, stride0=n8}.
// Caller (wave 0) later does s_wait_tensorcnt 0 before the workgroup barrier.
// ---------------------------------------------------------------------------
#if HAVE_TDM
__device__ inline void tdm_issue_1d(const half_t* gsrc, int n8) {
  const unsigned long long ga = (unsigned long long)(const void*)gsrc;
  v4u g0;
  g0[0] = 1u;                                             // count=1, no gather
  g0[1] = 0u;                                             // lds_addr = 0
  g0[2] = (unsigned)(ga & 0xFFFFFFFFu);                   // global_addr[31:0]
  g0[3] = (unsigned)((ga >> 32) & 0x1FFFFFFu) | (2u << 30); // addr[56:32]|type=2
  v8i_ g1;
  g1[0] = (int)(3u << 16);                                // data_size=3 (8B)
  g1[1] = (int)(((unsigned)n8 & 0xFFFFu) << 16);          // tensor_dim0 lo16
  g1[2] = (int)(((unsigned)n8 >> 16) | (1u << 16));       // dim0 hi16 | tensor_dim1=1
  g1[3] = (int)(((unsigned)n8 & 0xFFFFu) << 16);          // tile_dim0 (n8 < 65536)
  g1[4] = 1;                                              // tile_dim1=1
  g1[5] = n8;                                             // tensor_dim0_stride lo32
  g1[6] = 0;
  g1[7] = 0;
  v4i_ gz = {0, 0, 0, 0};
#if defined(__clang_major__) && __clang_major__ >= 23
  v8i_ gz8 = {0, 0, 0, 0, 0, 0, 0, 0};
  __builtin_amdgcn_tensor_load_to_lds(g0, g1, gz, gz, gz8, 0);
#else
  __builtin_amdgcn_tensor_load_to_lds(g0, g1, gz, gz, 0);
#endif
}
#endif

// ---------------------------------------------------------------------------
// Fused LSTM cell step:  g = x@Wih^T + h@Whh^T + b ; gates ; c,h update.
// Block = 128 threads (4 waves): BT=32 batch rows x 64 hidden cols.
// Each wave: 32x16 output tile, 4 gates x 2 row-groups = 8 f32 accumulators;
// every weight fragment feeds 2 WMMAs (B reuse). The 32xNH h tile (32KB,
// contiguous) is staged to LDS by the Tensor Data Mover, overlapped with the
// x@Wih GEMM from global. gridDim.z = #directions (weights/state strided).
// ---------------------------------------------------------------------------
__global__ __launch_bounds__(128) void lstm_cell_kernel(
    const half_t* __restrict__ xf, const half_t* __restrict__ xb, int ldx, int Kx,
    const half_t* __restrict__ Wih, size_t wih_ds,
    const half_t* __restrict__ Whh, size_t whh_ds,
    const float*  __restrict__ bias, size_t bias_ds,
    const half_t* __restrict__ h_in, float* __restrict__ c_st,
    half_t* __restrict__ h_out, size_t st_ds,
    half_t* __restrict__ yfp, half_t* __restrict__ ybp, int ldy)
{
  __shared__ half_t sh_h[BT * NH];   // 32 KB — single LDS object, offset 0

  const int d = blockIdx.z;
  const half_t* x   = d ? xb : xf;
  const half_t* Wi  = Wih + d * wih_ds;
  const half_t* Wh  = Whh + d * whh_ds;
  const float*  bs  = bias + d * bias_ds;
  const half_t* hin = h_in + d * st_ds;
  float*  cst  = c_st  + d * st_ds;
  half_t* hout = h_out + d * st_ds;
  half_t* yp   = d ? ybp : yfp;

  const int wave = threadIdx.x >> 5;
  const int n0 = blockIdx.x * 64 + wave * 16;   // hidden-column tile
  const int b0 = blockIdx.y * BT;               // batch-row tile (32 rows)

  // Kick off async staging of the recurrent A-operand (h rows b0..b0+31).
#if HAVE_TDM
  if (threadIdx.x < 32) tdm_issue_1d(hin + (size_t)b0 * NH, BT * NH * 2 / 8);
#else
  {
    const v8h* src = (const v8h*)(hin + (size_t)b0 * NH);
    v8h* dst = (v8h*)sh_h;
    for (int i = threadIdx.x; i < BT * NH / 8; i += 128) dst[i] = src[i];
  }
#endif

  v8f acc[4][2];
#pragma unroll
  for (int g = 0; g < 4; ++g)
#pragma unroll
    for (int rg = 0; rg < 2; ++rg) acc[g][rg] = {};

  // Input contribution (global), overlapped with the TDM transfer.
  for (int kc = 0; kc < Kx; kc += 32) {
    v16h a0 = load_a_frag(x, ldx, b0, kc);
    v16h a1 = load_a_frag(x, ldx, b0 + 16, kc);
#pragma unroll
    for (int g = 0; g < 4; ++g) {
      v16h w = load_b_frag(Wi, Kx, g * NH + n0, kc);
      acc[g][0] = WMMA_F16(a0, w, acc[g][0]);
      acc[g][1] = WMMA_F16(a1, w, acc[g][1]);
    }
  }

#if HAVE_TDM
  if (threadIdx.x < 32) __builtin_amdgcn_s_wait_tensorcnt(0);
#endif
  __syncthreads();

  // Recurrent contribution: A from LDS (ds_load), B streamed from L2.
  for (int kc = 0; kc < NH; kc += 32) {
    v16h a0 = load_a_frag(sh_h, NH, 0, kc);
    v16h a1 = load_a_frag(sh_h, NH, 16, kc);
#pragma unroll
    for (int g = 0; g < 4; ++g) {
      v16h w = load_b_frag(Wh, NH, g * NH + n0, kc);
      acc[g][0] = WMMA_F16(a0, w, acc[g][0]);
      acc[g][1] = WMMA_F16(a1, w, acc[g][1]);
    }
  }

  // Epilogue: C/D layout — VGPR r holds (row r, col lane&15) for lanes 0-15,
  // (row r+8, col lane-16) for lanes 16-31; row-group rg adds 16.
  const int lane = threadIdx.x & 31;
  const int n    = n0 + (lane & 15);
  const int mtop = (lane & 16) ? 8 : 0;
  const float bi = bs[0 * NH + n];
  const float bf = bs[1 * NH + n];
  const float bg = bs[2 * NH + n];
  const float bo = bs[3 * NH + n];
#pragma unroll
  for (int rg = 0; rg < 2; ++rg) {
#pragma unroll
    for (int r = 0; r < 8; ++r) {
      const int bm = b0 + rg * 16 + mtop + r;
      const float gi = sigm(acc[0][rg][r] + bi);
      const float gf = sigm(acc[1][rg][r] + bf);
      const float gg = tanhf(acc[2][rg][r] + bg);
      const float go = sigm(acc[3][rg][r] + bo);
      const size_t idx = (size_t)bm * NH + n;
      const float cn = gf * cst[idx] + gi * gg;
      cst[idx] = cn;
      const half_t hh = (half_t)(go * tanhf(cn));
      hout[idx] = hh;
      if (yp) yp[(size_t)bm * ldy + n] = hh;   // encoder layer-0 sequence output
    }
  }
}

// ---------------------------------------------------------------------------
// fc1: z = relu(h1 @ W^T + b), B x 256, K = 512 — same WMMA tiling (32x16).
// ---------------------------------------------------------------------------
__global__ __launch_bounds__(128) void fc1_kernel(
    const half_t* __restrict__ h, const half_t* __restrict__ W,
    const float* __restrict__ bias, half_t* __restrict__ z)
{
  const int wave = threadIdx.x >> 5;
  const int n0 = blockIdx.x * 64 + wave * 16;
  const int b0 = blockIdx.y * BT;
  v8f acc0 = {}, acc1 = {};
  for (int kc = 0; kc < NH; kc += 32) {
    v16h a0 = load_a_frag(h, NH, b0, kc);
    v16h a1 = load_a_frag(h, NH, b0 + 16, kc);
    v16h w  = load_b_frag(W, NH, n0, kc);
    acc0 = WMMA_F16(a0, w, acc0);
    acc1 = WMMA_F16(a1, w, acc1);
  }
  const int lane = threadIdx.x & 31;
  const int n    = n0 + (lane & 15);
  const int mtop = (lane & 16) ? 8 : 0;
  const float bb = bias[n];
#pragma unroll
  for (int r = 0; r < 8; ++r) {
    float v0 = acc0[r] + bb; v0 = v0 > 0.f ? v0 : 0.f;
    float v1 = acc1[r] + bb; v1 = v1 > 0.f ? v1 : 0.f;
    z[(size_t)(b0 + mtop + r) * NF + n]      = (half_t)v0;
    z[(size_t)(b0 + 16 + mtop + r) * NF + n] = (half_t)v1;
  }
}

// ---------------------------------------------------------------------------
// fc2 (256 -> 1 dot per batch row, wave reduction) + build next decoder input
// x_next = [pred, tf[:,p,1:8], 0-pad to 32].
// ---------------------------------------------------------------------------
__global__ __launch_bounds__(64) void fc2_next_kernel(
    const half_t* __restrict__ z, const float* __restrict__ w2,
    const float* __restrict__ b2, const float* __restrict__ tft,
    int p, float* __restrict__ out, half_t* __restrict__ xn)
{
  const int b    = blockIdx.x * 2 + (threadIdx.x >> 5);
  const int lane = threadIdx.x & 31;
  float s = 0.f;
#pragma unroll
  for (int k = lane; k < NF; k += 32) s += (float)z[(size_t)b * NF + k] * w2[k];
#pragma unroll
  for (int off = 16; off; off >>= 1) s += __shfl_xor(s, off, 32);
  const float pred = s + b2[0];
  if (lane == 0) {
    out[(size_t)b * NP + p] = pred;
    xn[(size_t)b * 32 + 0]  = (half_t)pred;
  } else if (lane < NI) {
    xn[(size_t)b * 32 + lane] = (half_t)tft[((size_t)b * NP + p) * NI + lane];
  } else {
    xn[(size_t)b * 32 + lane] = (half_t)0.f;
  }
}

// h_dec = fwd + bwd final hidden (f16), c_dec = fwd + bwd final cell (f32)
__global__ __launch_bounds__(256) void init_dec_state(
    const half_t* __restrict__ he, const float* __restrict__ ce,
    half_t* __restrict__ hd, float* __restrict__ cd)
{
  const int i = blockIdx.x * 256 + threadIdx.x;
  hd[i] = (half_t)((float)he[i] + (float)he[i + NB * NH]);
  cd[i] = ce[i] + ce[i + NB * NH];
}

// f32 [N][Ks] (row stride srcStride) -> f16 [N][Kd], zero-padded
__global__ void convert_pad_kernel(const float* __restrict__ src, int srcStride, int Ks,
                                   half_t* __restrict__ dst, int Kd, long total)
{
  const long i = (long)blockIdx.x * 256 + threadIdx.x;
  if (i >= total) return;
  const long n = i / Kd;
  const int  k = (int)(i % Kd);
  dst[i] = (k < Ks) ? (half_t)src[(size_t)n * srcStride + k] : (half_t)0.f;
}

// x f32 [B][S][I] -> f16 [S][B][32] zero-padded (time-major for the scan)
__global__ void convert_x_kernel(const float* __restrict__ x, half_t* __restrict__ xf)
{
  const long i = (long)blockIdx.x * 256 + threadIdx.x;
  if (i >= (long)NS * NB * 32) return;
  const int  k  = (int)(i & 31);
  const long bt = i >> 5;
  const int  b  = (int)(bt % NB);
  const int  t  = (int)(bt / NB);
  xf[i] = (k < NI) ? (half_t)x[((size_t)b * NS + t) * NI + k] : (half_t)0.f;
}

__global__ void zero_u32(unsigned int* __restrict__ p, long n)
{
  const long i = (long)blockIdx.x * 256 + threadIdx.x;
  if (i < n) p[i] = 0u;
}

// ---------------------------------------------------------------------------
extern "C" void kernel_launch(void* const* d_in, const int* in_sizes, int n_in,
                              void* d_out, int out_size, void* d_ws, size_t ws_size,
                              hipStream_t stream)
{
  (void)in_sizes; (void)n_in; (void)out_size; (void)ws_size;
  const float* x     = (const float*)d_in[0];
  const float* tft   = (const float*)d_in[1];
  const float* eWih0 = (const float*)d_in[2];
  const float* eWhh0 = (const float*)d_in[3];
  const float* eb0   = (const float*)d_in[4];
  const float* eWih1 = (const float*)d_in[5];
  const float* eWhh1 = (const float*)d_in[6];
  const float* eb1   = (const float*)d_in[7];
  const float* dWih0 = (const float*)d_in[8];
  const float* dWhh0 = (const float*)d_in[9];
  const float* db0   = (const float*)d_in[10];
  const float* dWih1 = (const float*)d_in[11];
  const float* dWhh1 = (const float*)d_in[12];
  const float* db1   = (const float*)d_in[13];
  const float* fc1W  = (const float*)d_in[14];
  const float* fc1b  = (const float*)d_in[15];
  const float* fc2W  = (const float*)d_in[16];
  const float* fc2b  = (const float*)d_in[17];
  float* out = (float*)d_out;

  char* ws = (char*)d_ws;
  size_t off = 0;
  auto take = [&](size_t bytes) -> char* {
    char* p = ws + off; off = (off + bytes + 255) & ~(size_t)255; return p;
  };
  half_t* xf16  = (half_t*)take((size_t)NS * NB * 32 * sizeof(half_t));
  half_t* y0    = (half_t*)take((size_t)NS * NB * (2 * NH) * sizeof(half_t));
  half_t* wih0  = (half_t*)take((size_t)2 * G4 * 32 * sizeof(half_t));
  half_t* whh0  = (half_t*)take((size_t)2 * G4 * NH * sizeof(half_t));
  half_t* wih1  = (half_t*)take((size_t)2 * G4 * (2 * NH) * sizeof(half_t));
  half_t* whh1  = (half_t*)take((size_t)2 * G4 * NH * sizeof(half_t));
  half_t* dwih0 = (half_t*)take((size_t)G4 * 32 * sizeof(half_t));
  half_t* dwhh0 = (half_t*)take((size_t)G4 * NH * sizeof(half_t));
  half_t* dwih1 = (half_t*)take((size_t)G4 * NH * sizeof(half_t));
  half_t* dwhh1 = (half_t*)take((size_t)G4 * NH * sizeof(half_t));
  half_t* wfc1  = (half_t*)take((size_t)NF * NH * sizeof(half_t));
  half_t* h0a   = (half_t*)take((size_t)2 * NB * NH * sizeof(half_t));
  half_t* h0b   = (half_t*)take((size_t)2 * NB * NH * sizeof(half_t));
  float*  c0    = (float*) take((size_t)2 * NB * NH * sizeof(float));
  half_t* h1a   = (half_t*)take((size_t)2 * NB * NH * sizeof(half_t));
  half_t* h1b   = (half_t*)take((size_t)2 * NB * NH * sizeof(half_t));
  float*  c1    = (float*) take((size_t)2 * NB * NH * sizeof(float));
  half_t* dh0a  = (half_t*)take((size_t)NB * NH * sizeof(half_t));
  half_t* dh0b  = (half_t*)take((size_t)NB * NH * sizeof(half_t));
  float*  dc0   = (float*) take((size_t)NB * NH * sizeof(float));
  half_t* dh1a  = (half_t*)take((size_t)NB * NH * sizeof(half_t));
  half_t* dh1b  = (half_t*)take((size_t)NB * NH * sizeof(half_t));
  float*  dc1   = (float*) take((size_t)NB * NH * sizeof(float));
  half_t* zbuf  = (half_t*)take((size_t)NB * NF * sizeof(half_t));
  half_t* dxbuf = (half_t*)take((size_t)NB * 32 * sizeof(half_t));

  // ---- weight / input conversions to f16 (weights become L2-resident) ----
  auto cvt = [&](const float* src, int Ks, half_t* dst, int Kd, long N) {
    const long total = N * Kd;
    convert_pad_kernel<<<(unsigned)((total + 255) / 256), 256, 0, stream>>>(src, Ks, Ks, dst, Kd, total);
  };
  cvt(eWih0, NI,     wih0,  32,     2L * G4);
  cvt(eWhh0, NH,     whh0,  NH,     2L * G4);
  cvt(eWih1, 2 * NH, wih1,  2 * NH, 2L * G4);
  cvt(eWhh1, NH,     whh1,  NH,     2L * G4);
  cvt(dWih0, NI,     dwih0, 32,     (long)G4);
  cvt(dWhh0, NH,     dwhh0, NH,     (long)G4);
  cvt(dWih1, NH,     dwih1, NH,     (long)G4);
  cvt(dWhh1, NH,     dwhh1, NH,     (long)G4);
  cvt(fc1W,  NH,     wfc1,  NH,     (long)NF);

  {
    const long total = (long)NS * NB * 32;
    convert_x_kernel<<<(unsigned)((total + 255) / 256), 256, 0, stream>>>(x, xf16);
  }
  // decoder input 0 = x[:, -1, :], zero-padded to 32
  convert_pad_kernel<<<(NB * 32 + 255) / 256, 256, 0, stream>>>(
      x + (size_t)(NS - 1) * NI, NS * NI, NI, dxbuf, 32, (long)NB * 32);

  // ---- zero initial states (harness poisons ws with 0xAA) ----
  {
    const long nzh = (long)2 * NB * NH / 2;   // halves -> u32
    const long nzc = (long)2 * NB * NH;
    zero_u32<<<(unsigned)((nzh + 255) / 256), 256, 0, stream>>>((unsigned int*)h0a, nzh);
    zero_u32<<<(unsigned)((nzh + 255) / 256), 256, 0, stream>>>((unsigned int*)h1a, nzh);
    zero_u32<<<(unsigned)((nzc + 255) / 256), 256, 0, stream>>>((unsigned int*)c0, nzc);
    zero_u32<<<(unsigned)((nzc + 255) / 256), 256, 0, stream>>>((unsigned int*)c1, nzc);
  }

  const dim3 cellGrid2(NH / 64, NB / BT, 2);
  const dim3 cellGrid1(NH / 64, NB / BT, 1);
  const dim3 fc1Grid(NF / 64, NB / BT, 1);

  // ---- encoder layer 0: fwd + bwd scans fused per step (gridDim.z = 2) ----
  for (int t = 0; t < NS; ++t) {
    const half_t* hin = (t & 1) ? h0b : h0a;
    half_t* hout      = (t & 1) ? h0a : h0b;
    lstm_cell_kernel<<<cellGrid2, 128, 0, stream>>>(
        xf16 + (size_t)t * NB * 32, xf16 + (size_t)(NS - 1 - t) * NB * 32, 32, 32,
        wih0, (size_t)G4 * 32, whh0, (size_t)G4 * NH, eb0, (size_t)G4,
        hin, c0, hout, (size_t)NB * NH,
        y0 + (size_t)t * NB * (2 * NH),
        y0 + (size_t)(NS - 1 - t) * NB * (2 * NH) + NH, 2 * NH);
  }
  half_t* h0fin = ((NS - 1) & 1) ? h0a : h0b;

  // ---- encoder layer 1 over y0 (Kx = 1024) ----
  for (int t = 0; t < NS; ++t) {
    const half_t* hin = (t & 1) ? h1b : h1a;
    half_t* hout      = (t & 1) ? h1a : h1b;
    lstm_cell_kernel<<<cellGrid2, 128, 0, stream>>>(
        y0 + (size_t)t * NB * (2 * NH), y0 + (size_t)(NS - 1 - t) * NB * (2 * NH),
        2 * NH, 2 * NH,
        wih1, (size_t)G4 * (2 * NH), whh1, (size_t)G4 * NH, eb1, (size_t)G4,
        hin, c1, hout, (size_t)NB * NH,
        nullptr, nullptr, 0);
  }
  half_t* h1fin = ((NS - 1) & 1) ? h1a : h1b;

  // ---- decoder initial states: sum of fwd/bwd finals ----
  init_dec_state<<<NB * NH / 256, 256, 0, stream>>>(h0fin, c0, dh0a, dc0);
  init_dec_state<<<NB * NH / 256, 256, 0, stream>>>(h1fin, c1, dh1a, dc1);

  // ---- decoder: 24 autoregressive steps ----
  for (int p = 0; p < NP; ++p) {
    half_t* h0cur = (p & 1) ? dh0b : dh0a;
    half_t* h0nxt = (p & 1) ? dh0a : dh0b;
    half_t* h1cur = (p & 1) ? dh1b : dh1a;
    half_t* h1nxt = (p & 1) ? dh1a : dh1b;
    lstm_cell_kernel<<<cellGrid1, 128, 0, stream>>>(
        dxbuf, dxbuf, 32, 32,
        dwih0, 0, dwhh0, 0, db0, 0,
        h0cur, dc0, h0nxt, 0, nullptr, nullptr, 0);
    lstm_cell_kernel<<<cellGrid1, 128, 0, stream>>>(
        h0nxt, h0nxt, NH, NH,
        dwih1, 0, dwhh1, 0, db1, 0,
        h1cur, dc1, h1nxt, 0, nullptr, nullptr, 0);
    fc1_kernel<<<fc1Grid, 128, 0, stream>>>(h1nxt, wfc1, fc1b, zbuf);
    fc2_next_kernel<<<NB / 2, 64, 0, stream>>>(zbuf, fc2W, fc2b, tft, p, out, dxbuf);
  }
}
